// AdaptiveWeight_23527830847988
// MI455X (gfx1250) — matrix operand order repaired and checked by reading, not compiled, not run on hardware
//
#include <hip/hip_runtime.h>

// Problem constants (from reference)
#define B_    2
#define N1_   4096
#define N2_   4096
#define C_    256
#define NS_   32
#define RAD   0.1f
#define R2_   (RAD*RAD)
#define EPS_  1e-5f

typedef __attribute__((ext_vector_type(16))) _Float16 v16h;
typedef __attribute__((ext_vector_type(8)))  float    v8f;
typedef __attribute__((ext_vector_type(8)))  int      v8i;

// ------------------------------------------------------------------
// Kernel 0: weight re-layout.
//  w1B  : w1^T in B-matrix (32x16) fragment order for kernel 2:
//         elem[((ot*8+kc)*32+lane)*16 + e] = w1[ot*16+(lane&15)][kc*32 + 8*(lane>>4) + e + 8*(e>=8)]
//  woutA: w_out in A-matrix (16x32) fragment order for kernel 3:
//         K = kc*32 + 16*(lane>>4) + e (contiguous per lane)
// ------------------------------------------------------------------
__global__ void prep_weights(const float* __restrict__ w1,
                             const float* __restrict__ wout,
                             _Float16* __restrict__ w1B,
                             _Float16* __restrict__ woutA) {
  int e    = blockIdx.x * blockDim.x + threadIdx.x;   // 0..65535
  int lane = (e >> 4) & 31;
  int kc   = (e >> 9) & 7;
  int ot   = e >> 12;
  int m    = lane & 15;
  int hi   = lane >> 4;
  int o    = ot * 16 + m;

  // B-layout element (table K order: 8*hi + el, +8 for el >= 8)
  int el  = e & 15;
  int cB  = kc * 32 + 8 * hi + el + ((el >= 8) ? 8 : 0);
  w1B[e]  = (_Float16)w1[o * C_ + cB];

  // A-layout element (contiguous K per lane)
  int t   = e & 1;
  int j   = (e >> 1) & 7;
  int KA  = 16 * hi + el;
  int cA  = kc * 32 + KA;
  woutA[e] = (_Float16)wout[o * C_ + cA];
}

// ------------------------------------------------------------------
// Kernel 1: masked ball query — first NS_ in-radius support indices
// (ascending index order == reference's top_k of -arange).
// ------------------------------------------------------------------
__global__ __launch_bounds__(256)
void ball_query(const float* __restrict__ qxyz,
                const float* __restrict__ sxyz,
                const float* __restrict__ smask,
                int* __restrict__ idxBuf,
                int* __restrict__ cntBuf) {
  __shared__ float sx[1024], sy[1024], sz[1024], sm[1024];
  int qid = blockIdx.x * blockDim.x + threadIdx.x;   // 0..8191
  int b   = qid >> 12;
  float qx = qxyz[qid * 3 + 0];
  float qy = qxyz[qid * 3 + 1];
  float qz = qxyz[qid * 3 + 2];
  int cnt = 0;
  int sb  = b * N2_;
  for (int tile = 0; tile < N2_; tile += 1024) {
    __syncthreads();
    for (int i = threadIdx.x; i < 1024; i += blockDim.x) {
      int s = sb + tile + i;
      sx[i] = sxyz[s * 3 + 0];
      sy[i] = sxyz[s * 3 + 1];
      sz[i] = sxyz[s * 3 + 2];
      sm[i] = smask[s];
    }
    __syncthreads();
    for (int jj = 0; jj < 1024; ++jj) {
      float dx = sx[jj] - qx, dy = sy[jj] - qy, dz = sz[jj] - qz;
      float d2 = dx * dx + dy * dy + dz * dz;
      if (d2 < R2_ && sm[jj] > 0.0f && cnt < NS_) {
        idxBuf[qid * NS_ + cnt] = tile + jj;
        cnt++;
      }
    }
  }
  cntBuf[qid] = cnt;
  for (int k = cnt; k < NS_; ++k) idxBuf[qid * NS_ + k] = 0;
}

// ------------------------------------------------------------------
// Kernel 2: per-query fused  h = relu(w0*rel+b0);
//   cw^T = h^T * w1^T via WMMA (samples in M/register dim, o in lanes);
//   agg[o] = sum_s cw*gfeat*mask / denom   -> aggBuf (f16 rows of 256).
//
// Loop nest: ot ROLLED, kc unrolled, both s-tiles consume each B-frag
// immediately (B loaded once, two WMMAs). A-frags for both s-tiles are
// held in registers (128 VGPRs, fixed). No register arrays escape a
// rolled loop -> no scratch spills, no VGPR-MSB addressing.
// ------------------------------------------------------------------
#define WAVES2 2
#define HS_ 264   // padded LDS row stride in halves (528 B, 16B-aligned)

__global__ __launch_bounds__(64)
void adaptive_main(const float* __restrict__ qxyz,
                   const float* __restrict__ sxyz,
                   const float* __restrict__ qmask,
                   const float* __restrict__ feats,
                   const float* __restrict__ w0,
                   const float* __restrict__ b0,
                   const float* __restrict__ b1,
                   const int* __restrict__ idxBuf,
                   const int* __restrict__ cntBuf,
                   const _Float16* __restrict__ w1B,
                   _Float16* __restrict__ aggBuf) {
  extern __shared__ __align__(16) char smem[];
  _Float16* hls  = (_Float16*)smem;                        // WAVES2*32*HS_ halves
  int*      ilds = (int*)(smem + WAVES2 * 32 * HS_ * 2);   // WAVES2*32

  int tid  = threadIdx.x;
  int wave = tid >> 5;
  int lane = tid & 31;
  int q    = blockIdx.x * WAVES2 + wave;   // 0..8191
  int b    = q >> 12;

  int   cnt   = cntBuf[q];
  int   myidx = idxBuf[q * NS_ + lane];
  float qm    = qmask[q];
  float omq   = 1.0f - qm;                 // (1 - query_mask) mask term
  ilds[wave * 32 + lane] = myidx;
  float denom = (float)cnt + 32.0f * omq;

  // relative position for my sample (lane == s)
  float qx = qxyz[q * 3 + 0], qy = qxyz[q * 3 + 1], qz = qxyz[q * 3 + 2];
  int   sb = (b * N2_ + myidx) * 3;
  float rx = (sxyz[sb + 0] - qx) * (1.0f / RAD);
  float ry = (sxyz[sb + 1] - qy) * (1.0f / RAD);
  float rz = (sxyz[sb + 2] - qz) * (1.0f / RAD);

  // h row (256 f16) for my sample -> LDS (rows = samples)
  _Float16* hrow = hls + (wave * 32 + lane) * HS_;
  for (int c = 0; c < C_; c += 2) {
    float h0 = fmaxf(w0[c * 3 + 0] * rx + w0[c * 3 + 1] * ry + w0[c * 3 + 2] * rz + b0[c], 0.0f);
    float h1 = fmaxf(w0[c * 3 + 3] * rx + w0[c * 3 + 4] * ry + w0[c * 3 + 5] * rz + b0[c + 1], 0.0f);
    unsigned u0 = (unsigned)__builtin_bit_cast(unsigned short, (_Float16)h0);
    unsigned u1 = (unsigned)__builtin_bit_cast(unsigned short, (_Float16)h1);
    *(unsigned*)(hrow + c) = u0 | (u1 << 16);
  }
  __syncthreads();

  int nloc = lane & 15;
  int hi   = lane >> 4;

  // A fragments (h^T) for both s-tiles, held in registers.
  // A-layout element e: K = e + 8*((e>=8)+hi) -> two contiguous 16B chunks.
  const _Float16* hrA0 = hls + (wave * 32 + nloc) * HS_;        // s-tile 0
  const _Float16* hrA1 = hls + (wave * 32 + 16 + nloc) * HS_;   // s-tile 1
  v16h af0[8], af1[8];
#pragma unroll
  for (int kc = 0; kc < 8; ++kc) {
    uint4 lo0 = *(const uint4*)(hrA0 + kc * 32 + 8 * hi);
    uint4 hq0 = *(const uint4*)(hrA0 + kc * 32 + 16 + 8 * hi);
    uint4 lo1 = *(const uint4*)(hrA1 + kc * 32 + 8 * hi);
    uint4 hq1 = *(const uint4*)(hrA1 + kc * 32 + 16 + 8 * hi);
    v8i p0, p1;
    p0[0] = lo0.x; p0[1] = lo0.y; p0[2] = lo0.z; p0[3] = lo0.w;
    p0[4] = hq0.x; p0[5] = hq0.y; p0[6] = hq0.z; p0[7] = hq0.w;
    p1[0] = lo1.x; p1[1] = lo1.y; p1[2] = lo1.z; p1[3] = lo1.w;
    p1[4] = hq1.x; p1[5] = hq1.y; p1[6] = hq1.z; p1[7] = hq1.w;
    af0[kc] = __builtin_bit_cast(v16h, p0);
    af1[kc] = __builtin_bit_cast(v16h, p1);
  }

  // Neighbor indices for epilogue: D-layout M = r + 8*hi -> sample.
  // st0: s = 8*hi + r ; st1: s = 16 + 8*hi + r   (r = 0..7)
  int4 ia0 = *(const int4*)(ilds + wave * 32 + 8 * hi);
  int4 ib0 = *(const int4*)(ilds + wave * 32 + 8 * hi + 4);
  int4 ia1 = *(const int4*)(ilds + wave * 32 + 16 + 8 * hi);
  int4 ib1 = *(const int4*)(ilds + wave * 32 + 16 + 8 * hi + 4);
  int idx0[8] = {ia0.x, ia0.y, ia0.z, ia0.w, ib0.x, ib0.y, ib0.z, ib0.w};
  int idx1[8] = {ia1.x, ia1.y, ia1.z, ia1.w, ib1.x, ib1.y, ib1.z, ib1.w};

  float inv = 1.0f / denom;
  const float* fbase = feats + (size_t)b * C_ * N2_;

  for (int ot = 0; ot < 16; ++ot) {          // rolled: keeps live set small
    v8f acc0 = {}, acc1 = {};
#pragma unroll
    for (int kc = 0; kc < 8; ++kc) {
      v16h bfr = *(const v16h*)(w1B + (((ot * 8 + kc) * 32 + lane) << 4));
      acc0 = __builtin_amdgcn_wmma_f32_16x16x32_f16(
          false, af0[kc], false, bfr, (short)0, acc0, false, false);
      acc1 = __builtin_amdgcn_wmma_f32_16x16x32_f16(
          false, af1[kc], false, bfr, (short)0, acc1, false, false);
    }
    int o = ot * 16 + nloc;                  // D-layout: N = lane&15 -> o
    float b1o = b1[o];
    const float* frow = fbase + (size_t)o * N2_;
    float tot = 0.0f;
#pragma unroll
    for (int r = 0; r < 8; ++r) {            // D-layout: M = r+8*hi -> sample
      int   s0 = 8 * hi + r;
      int   s1 = 16 + s0;
      float m0 = ((s0 < cnt) ? 1.0f : 0.0f) + omq;
      float m1 = ((s1 < cnt) ? 1.0f : 0.0f) + omq;
      tot += (acc0[r] + b1o) * frow[idx0[r]] * m0;
      tot += (acc1[r] + b1o) * frow[idx1[r]] * m1;
    }
    tot += __shfl_xor(tot, 16, 32);          // combine the two M halves
    if (hi == 0)
      aggBuf[q * C_ + o] = (_Float16)(tot * inv);
  }
}

// ------------------------------------------------------------------
// Kernel 3: out = relu(BN(w_out @ agg)) via WMMA. One wave per 16
// columns (query points); B fragments read straight from aggBuf.
// ------------------------------------------------------------------
__global__ __launch_bounds__(256)
void out_gemm_bn(const _Float16* __restrict__ woutA,
                 const _Float16* __restrict__ aggBuf,
                 const float* __restrict__ gamma,
                 const float* __restrict__ beta,
                 const float* __restrict__ mean,
                 const float* __restrict__ var,
                 float* __restrict__ out) {
  int tid   = threadIdx.x;
  int wave  = tid >> 5;
  int lane  = tid & 31;
  int tile  = blockIdx.x * 8 + wave;   // 0..511
  int nloc  = lane & 15;
  int hi    = lane >> 4;
  int kbase = hi * 16;
  int q     = tile * 16 + nloc;        // flat column (b*N1 + n)
  int b     = q >> 12;
  int nq    = q & (N1_ - 1);

  v16h bf[8];
#pragma unroll
  for (int kc = 0; kc < 8; ++kc)
    bf[kc] = *(const v16h*)(aggBuf + q * C_ + kc * 32 + kbase);

  for (int ot = 0; ot < 16; ++ot) {
    v8f acc = {};
#pragma unroll
    for (int kc = 0; kc < 8; ++kc) {
      v16h af = *(const v16h*)(woutA + (((ot * 8 + kc) * 32 + lane) << 4));
      acc = __builtin_amdgcn_wmma_f32_16x16x32_f16(
          false, af, false, bf[kc], (short)0, acc, false, false);
    }
#pragma unroll
    for (int r = 0; r < 8; ++r) {
      int   o  = ot * 16 + hi * 8 + r;
      float sc = gamma[o] * rsqrtf(var[o] + EPS_);
      float v  = (acc[r] - mean[o]) * sc + beta[o];
      out[(b * C_ + o) * N1_ + nq] = fmaxf(v, 0.0f);
    }
  }
}

// ------------------------------------------------------------------
extern "C" void kernel_launch(void* const* d_in, const int* in_sizes, int n_in,
                              void* d_out, int out_size, void* d_ws, size_t ws_size,
                              hipStream_t stream) {
  const float* qxyz  = (const float*)d_in[0];
  const float* sxyz  = (const float*)d_in[1];
  const float* qmask = (const float*)d_in[2];
  const float* smask = (const float*)d_in[3];
  const float* feats = (const float*)d_in[4];
  const float* w0    = (const float*)d_in[5];
  const float* b0    = (const float*)d_in[6];
  const float* w1    = (const float*)d_in[7];
  const float* b1    = (const float*)d_in[8];
  const float* wout  = (const float*)d_in[9];
  const float* gamma = (const float*)d_in[10];
  const float* beta  = (const float*)d_in[11];
  const float* mean  = (const float*)d_in[12];
  const float* var   = (const float*)d_in[13];
  float* out = (float*)d_out;

  // Workspace carve (all offsets 256B aligned)
  char* ws = (char*)d_ws;
  int*       idxBuf = (int*)(ws + 0);                       // 8192*32 ints   = 1,048,576 B
  int*       cntBuf = (int*)(ws + 1048576);                 // 8192 ints      =    32,768 B
  _Float16*  w1B    = (_Float16*)(ws + 1081344);            // 65536 f16      =   131,072 B
  _Float16*  woutA  = (_Float16*)(ws + 1212416);            // 65536 f16      =   131,072 B
  _Float16*  aggBuf = (_Float16*)(ws + 1343488);            // 8192*256 f16   = 4,194,304 B

  prep_weights<<<256, 256, 0, stream>>>(w1, wout, w1B, woutA);
  ball_query<<<32, 256, 0, stream>>>(qxyz, sxyz, smask, idxBuf, cntBuf);

  size_t smem2 = (size_t)WAVES2 * 32 * HS_ * 2   // h staging
               + (size_t)WAVES2 * 32 * 4;        // idx
  adaptive_main<<<(B_ * N1_) / WAVES2, 32 * WAVES2, smem2, stream>>>(
      qxyz, sxyz, qmask, feats, w0, b0, b1, idxBuf, cntBuf, w1B, aggBuf);

  out_gemm_bn<<<64, 256, 0, stream>>>(woutA, aggBuf, gamma, beta, mean, var, out);
}
